// Attention_62998580298271
// MI455X (gfx1250) — compile-verified, matrix operand order checked by reading, not run on hardware
//
#include <hip/hip_runtime.h>

// ---------------------------------------------------------------------------
// ViT attention block for gfx1250 (CDNA5, wave32, WMMA).
//   B=8, N=1024, C=768, heads=12, head_dim=64
// Pipeline: qkv GEMM (f32->bf16, 16x64 blocked) ->
//           flash attention (bf16 WMMA, async K/V staging to LDS) ->
//           proj GEMM (+bias, 16x64 blocked, f32 out)
// ---------------------------------------------------------------------------

#define DIM   768
#define NH    12
#define HD    64
#define BATCH 8
#define SEQ   1024
#define SM_SCALE 0.125f   // 1/sqrt(64)

typedef __attribute__((ext_vector_type(16))) __bf16 v16bf;
typedef __attribute__((ext_vector_type(8)))  float  v8f;

union FragBF {
  unsigned       u32[8];
  unsigned short u16[16];
  v16bf          bf;
};

// K-offset held by fragment VGPR v for this lane-half (ISA 16-bit A/B layout):
// VGPR 0..3 -> K = 2v + 8*half ; VGPR 4..7 -> K = 16 + 2(v-4) + 8*half
__device__ __forceinline__ int kbase(int v, int hf) {
  return ((v & 4) << 2) + ((v & 3) << 1) + (hf << 3);
}

// f32 -> bf16 (round-to-nearest-even)
__device__ __forceinline__ unsigned short f2bf(float f) {
  unsigned x = __builtin_bit_cast(unsigned, f);
  return (unsigned short)((x + 0x7fffu + ((x >> 16) & 1u)) >> 16);
}

// 16x32 A-fragment (or 32x16 B-fragment of row-major W) from bf16 matrix:
// this lane reads row (lane%16); `row` points at that row + chunk base.
__device__ __forceinline__ v16bf frag_bf16(const unsigned short* row, int hf) {
  FragBF f;
#pragma unroll
  for (int v = 0; v < 8; ++v)
    f.u32[v] = *(const unsigned*)(row + kbase(v, hf));
  return f.bf;
}

// Same, but f32 source: load pairs, convert to bf16.
__device__ __forceinline__ v16bf frag_f32(const float* row, int hf) {
  FragBF f;
#pragma unroll
  for (int v = 0; v < 8; ++v) {
    const float2 d = *(const float2*)(row + kbase(v, hf));
    f.u16[2 * v]     = f2bf(d.x);
    f.u16[2 * v + 1] = f2bf(d.y);
  }
  return f.bf;
}

__device__ __forceinline__ v8f wmma_bf16(v16bf a, v16bf b, v8f c) {
  return __builtin_amdgcn_wmma_f32_16x16x32_bf16(false, a, false, b, (short)0, c,
                                                 false, false);
}

// One 16-byte async copy global -> LDS (tracked by ASYNCcnt).
__device__ __forceinline__ void async_copy16(const unsigned short* gsrc,
                                             unsigned ldsoff) {
  asm volatile("global_load_async_to_lds_b128 %0, %1, off"
               :: "v"(ldsoff), "v"((unsigned long long)(size_t)gsrc)
               : "memory");
}

// ---------------------------------------------------------------------------
// Kernel 1: qkv = x @ w_qkv^T  (x:[8192,768] f32, w:[2304,768] f32)
// One wave -> one 16x64 output block (4 WMMA accumulators; A-frag reused 4x).
// 64-col block == exactly one (s, head). Stores bf16:
//   s=0 -> Q[b,h,n,d], s=1 -> K[b,h,n,d], s=2 -> Vt[b,h,d,n] (transposed)
// ---------------------------------------------------------------------------
__global__ void __launch_bounds__(256)
qkv_kernel(const float* __restrict__ x, const float* __restrict__ wqkv,
           unsigned short* __restrict__ q, unsigned short* __restrict__ k,
           unsigned short* __restrict__ vt) {
  const int lane = threadIdx.x & 31;
  const int tr = lane & 15, hf = lane >> 4;
  const int wave = blockIdx.x * (blockDim.x >> 5) + (threadIdx.x >> 5);
  const int NT = (3 * DIM) / 64;                 // 36 column blocks
  const int mt = wave / NT, nb = wave % NT;

  const float* xrow = x + (size_t)(mt * 16 + tr) * DIM;
  const float* wrow = wqkv + (size_t)(nb * 64 + tr) * DIM;

  v8f acc[4];
#pragma unroll
  for (int s = 0; s < 4; ++s) acc[s] = (v8f){0.f,0.f,0.f,0.f,0.f,0.f,0.f,0.f};

  for (int c = 0; c < DIM; c += 32) {
    if (c + 64 < DIM) __builtin_prefetch(xrow + c + 64, 0, 0);
    const v16bf a = frag_f32(xrow + c, hf);
#pragma unroll
    for (int sub = 0; sub < 4; ++sub)
      acc[sub] = wmma_bf16(a, frag_f32(wrow + (size_t)sub * 16 * DIM + c, hf),
                           acc[sub]);
  }

  const int ob   = nb * 64;                      // first output column
  const int s    = ob / DIM;                     // 0:q 1:k 2:v
  const int head = (ob - s * DIM) / HD;          // block aligns with one head
  const int row0 = mt * 16;
  const int b    = row0 / SEQ;
  const int bh   = b * NH + head;
#pragma unroll
  for (int sub = 0; sub < 4; ++sub) {
    const int dd = sub * 16 + tr;
#pragma unroll
    for (int r = 0; r < 8; ++r) {
      const int nseq = row0 - b * SEQ + r + 8 * hf;
      const unsigned short val = f2bf(acc[sub][r]);
      if (s == 0)      q [((size_t)bh * SEQ + nseq) * HD + dd] = val;
      else if (s == 1) k [((size_t)bh * SEQ + nseq) * HD + dd] = val;
      else             vt[((size_t)bh * HD + dd) * SEQ + nseq] = val;
    }
  }
}

// ---------------------------------------------------------------------------
// Kernel 2: flash attention. Block = 8 waves sharing one (b,head); each wave
// owns a 16-row query tile. Per 32-key block: K (32x64) and Vt (64x32) tiles
// are staged once per workgroup via global_load_async_to_lds_b128 (double
// buffered, ASYNCcnt + barrier), then S = Q K^T (WMMA from LDS), online
// softmax across 16-lane halves, P -> LDS -> A-frag, O += P @ V (4 WMMAs).
// ---------------------------------------------------------------------------
__global__ void __launch_bounds__(256)
attn_kernel(const unsigned short* __restrict__ q,
            const unsigned short* __restrict__ k,
            const unsigned short* __restrict__ vt,
            unsigned short* __restrict__ ao) {
  __shared__ unsigned short kbuf[2][32 * HD];    // 2 x 4KB : 32 keys x 64 d
  __shared__ unsigned short vbuf[2][HD * 32];    // 2 x 4KB : 64 d   x 32 keys
  __shared__ unsigned short pbuf[8][16 * 32];    // per-wave P tile (bf16)

  const int tid  = threadIdx.x;
  const int lane = tid & 31;
  const int tr = lane & 15, hf = lane >> 4;
  const int wv = tid >> 5;
  const int QT = SEQ / 16;                       // 64 query tiles per (b,h)
  const int wave0 = blockIdx.x * 8;
  const int bh = wave0 / QT;                     // uniform across the block
  const int qt = (wave0 % QT) + wv;
  const int b = bh / NH, head = bh % NH;

  const unsigned short* kg = k + (size_t)bh * SEQ * HD;
  const unsigned short* vg = vt + (size_t)bh * HD * SEQ;

  // Per-thread 16B staging slot: K tile = 32 rows x 8 chunks,
  //                              V tile = 64 rows x 4 chunks.
  const int krow = tid >> 3, kch = tid & 7;
  const int vrow = tid >> 2, vch = tid & 3;

  // Q fragments (16 rows x K=64 -> two 32-wide chunks), loaded once.
  const unsigned short* qrow = q + ((size_t)bh * SEQ + qt * 16 + tr) * HD;
  const v16bf aq0 = frag_bf16(qrow, hf);
  const v16bf aq1 = frag_bf16(qrow + 32, hf);

  v8f o0 = {0.f,0.f,0.f,0.f,0.f,0.f,0.f,0.f};
  v8f o1 = o0, o2 = o0, o3 = o0;
  float m_i[8], l_i[8];
#pragma unroll
  for (int r = 0; r < 8; ++r) { m_i[r] = -3.0e38f; l_i[r] = 0.f; }

  unsigned short* pl = pbuf[wv];

  for (int it = 0; it < SEQ / 32; ++it) {
    const int kb0 = it * 32;
    const int p   = it & 1;

    // --- cooperative async staging of K and Vt tiles into LDS ---
    async_copy16(kg + (size_t)(kb0 + krow) * HD + kch * 8,
                 (unsigned)(size_t)&kbuf[p][krow * HD + kch * 8]);
    async_copy16(vg + (size_t)vrow * SEQ + kb0 + vch * 8,
                 (unsigned)(size_t)&vbuf[p][vrow * 32 + vch * 8]);
    asm volatile("s_wait_asynccnt 0" ::: "memory");
    __syncthreads();

    // --- S tiles: 16 queries x 32 keys (fragments from LDS) ---
    v8f s0 = {0.f,0.f,0.f,0.f,0.f,0.f,0.f,0.f};
    v8f s1 = s0;
    const unsigned short* kr0 = &kbuf[p][tr * HD];
    s0 = wmma_bf16(aq0, frag_bf16(kr0, hf), s0);
    s0 = wmma_bf16(aq1, frag_bf16(kr0 + 32, hf), s0);
    const unsigned short* kr1 = &kbuf[p][(16 + tr) * HD];
    s1 = wmma_bf16(aq0, frag_bf16(kr1, hf), s1);
    s1 = wmma_bf16(aq1, frag_bf16(kr1 + 32, hf), s1);

    // --- online softmax; D-layout: acc[r] is element (r + 8*hf, tr) ---
#pragma unroll
    for (int r = 0; r < 8; ++r) {
      float v0 = s0[r] * SM_SCALE, v1 = s1[r] * SM_SCALE;
      float mx = fmaxf(v0, v1);
#pragma unroll
      for (int off = 1; off < 16; off <<= 1)
        mx = fmaxf(mx, __shfl_xor(mx, off, 16));
      const float mn = fmaxf(m_i[r], mx);
      const float al = __expf(m_i[r] - mn);
      const float p0 = __expf(v0 - mn);
      const float p1 = __expf(v1 - mn);
      float ps = p0 + p1;
#pragma unroll
      for (int off = 1; off < 16; off <<= 1)
        ps += __shfl_xor(ps, off, 16);
      l_i[r] = l_i[r] * al + ps;
      m_i[r] = mn;
      o0[r] = o0[r] * al; o1[r] = o1[r] * al;
      o2[r] = o2[r] * al; o3[r] = o3[r] * al;
      const int m = r + 8 * hf;
      pl[m * 32 + tr]      = f2bf(p0);
      pl[m * 32 + 16 + tr] = f2bf(p1);
    }
    asm volatile("s_wait_dscnt 0" ::: "memory");   // P stores visible to frag

    // --- O += P @ V : P 16x32 A-frag from LDS; Vt rows give B-frags ---
    const v16bf pf = frag_bf16(pl + tr * 32, hf);
    o0 = wmma_bf16(pf, frag_bf16(&vbuf[p][tr * 32], hf), o0);
    o1 = wmma_bf16(pf, frag_bf16(&vbuf[p][(16 + tr) * 32], hf), o1);
    o2 = wmma_bf16(pf, frag_bf16(&vbuf[p][(32 + tr) * 32], hf), o2);
    o3 = wmma_bf16(pf, frag_bf16(&vbuf[p][(48 + tr) * 32], hf), o3);
  }

  // epilogue: divide by softmax denominator, store bf16 [B,N,C]
#pragma unroll
  for (int r = 0; r < 8; ++r) {
    const float inv = 1.f / l_i[r];
    const int rowg = qt * 16 + r + 8 * hf;
    unsigned short* orow = ao + ((size_t)b * SEQ + rowg) * DIM + head * HD + tr;
    orow[0]  = f2bf(o0[r] * inv);
    orow[16] = f2bf(o1[r] * inv);
    orow[32] = f2bf(o2[r] * inv);
    orow[48] = f2bf(o3[r] * inv);
  }
}

// ---------------------------------------------------------------------------
// Kernel 3: out = ao @ w_proj^T + b_proj  (ao bf16 [8192,768] -> f32 out)
// One wave -> one 16x64 output block (4 accumulators, A-frag reused 4x).
// ---------------------------------------------------------------------------
__global__ void __launch_bounds__(256)
proj_kernel(const unsigned short* __restrict__ ao, const float* __restrict__ wp,
            const float* __restrict__ bias, float* __restrict__ out) {
  const int lane = threadIdx.x & 31;
  const int tr = lane & 15, hf = lane >> 4;
  const int wave = blockIdx.x * (blockDim.x >> 5) + (threadIdx.x >> 5);
  const int NT = DIM / 64;                       // 12 column blocks
  const int mt = wave / NT, nb = wave % NT;

  const unsigned short* arow = ao + (size_t)(mt * 16 + tr) * DIM;
  const float* wrow = wp + (size_t)(nb * 64 + tr) * DIM;

  v8f acc[4];
#pragma unroll
  for (int s = 0; s < 4; ++s) acc[s] = (v8f){0.f,0.f,0.f,0.f,0.f,0.f,0.f,0.f};

  for (int c = 0; c < DIM; c += 32) {
    if (c + 64 < DIM) __builtin_prefetch(arow + c + 64, 0, 0);
    const v16bf a = frag_bf16(arow + c, hf);
#pragma unroll
    for (int sub = 0; sub < 4; ++sub)
      acc[sub] = wmma_bf16(a, frag_f32(wrow + (size_t)sub * 16 * DIM + c, hf),
                           acc[sub]);
  }

#pragma unroll
  for (int sub = 0; sub < 4; ++sub) {
    const int col = nb * 64 + sub * 16 + tr;
    const float bv = bias[col];
#pragma unroll
    for (int r = 0; r < 8; ++r)
      out[(size_t)(mt * 16 + r + 8 * hf) * DIM + col] = acc[sub][r] + bv;
  }
}

// ---------------------------------------------------------------------------
extern "C" void kernel_launch(void* const* d_in, const int* in_sizes, int n_in,
                              void* d_out, int out_size, void* d_ws, size_t ws_size,
                              hipStream_t stream) {
  const float* x     = (const float*)d_in[0];
  const float* wqkv  = (const float*)d_in[1];
  const float* wproj = (const float*)d_in[2];
  const float* bproj = (const float*)d_in[3];
  float* out = (float*)d_out;

  // bf16 workspace: Q, K, Vt, attn-out  (4 x 12.6 MB = 48 MB; L2-resident)
  const size_t per = (size_t)BATCH * NH * SEQ * HD;   // 6,291,456 elems
  unsigned short* q  = (unsigned short*)d_ws;
  unsigned short* k  = q + per;
  unsigned short* vt = k + per;
  unsigned short* ao = vt + per;

  {
    const int waves = (BATCH * SEQ / 16) * (3 * DIM / 64);   // 512*36
    qkv_kernel<<<waves / 8, 256, 0, stream>>>(x, wqkv, q, k, vt);
  }
  {
    const int waves = BATCH * NH * (SEQ / 16);               // 6144
    attn_kernel<<<waves / 8, 256, 0, stream>>>(q, k, vt, ao);
  }
  {
    const int waves = (BATCH * SEQ / 16) * (DIM / 64);       // 512*12
    proj_kernel<<<waves / 8, 256, 0, stream>>>(ao, wproj, bproj, out);
  }
  (void)in_sizes; (void)n_in; (void)out_size; (void)ws_size;
}
